// GCNResNetConcatModel_89481348645024
// MI455X (gfx1250) — matrix operand order repaired and checked by reading, not compiled, not run on hardware
//
#include <hip/hip_runtime.h>
#include <hip/hip_bf16.h>
#include <stdint.h>
#include <stddef.h>

typedef __attribute__((ext_vector_type(16))) __bf16 v16bf;
typedef __attribute__((ext_vector_type(8)))  __bf16 bf16x8;
typedef __attribute__((ext_vector_type(8)))  float  v8f;
typedef __attribute__((ext_vector_type(4)))  unsigned int u32x4;
typedef __attribute__((ext_vector_type(8)))  int i32x8;
typedef __attribute__((ext_vector_type(4)))  int i32x4;
typedef __attribute__((ext_vector_type(4)))  int v4i;

// ---------------------------------------------------------------------------
// Elementwise / conversion kernels
// ---------------------------------------------------------------------------
__global__ void k_fill_f32(float* __restrict__ p, size_t n, float v) {
  size_t i = blockIdx.x * (size_t)blockDim.x + threadIdx.x;
  size_t st = (size_t)gridDim.x * blockDim.x;
  for (; i < n; i += st) p[i] = v;
}

__global__ void k_f32_to_bf16(const float* __restrict__ in, __bf16* __restrict__ out, size_t n) {
  size_t i = blockIdx.x * (size_t)blockDim.x + threadIdx.x;
  size_t st = (size_t)gridDim.x * blockDim.x;
  for (; i < n; i += st) out[i] = (__bf16)in[i];
}

// OIHW f32 weights -> [O][Kpad] bf16 (K = I*KH*KW); zero pad K tail.
// Flattened OIHW row-major IS the N x K (B-transposed) layout WMMA wants.
__global__ void k_w_pad_bf16(const float* __restrict__ w, __bf16* __restrict__ out,
                             int O, int K, int Kpad) {
  size_t total = (size_t)O * Kpad;
  size_t i = blockIdx.x * (size_t)blockDim.x + threadIdx.x;
  size_t st = (size_t)gridDim.x * blockDim.x;
  for (; i < total; i += st) {
    int o = (int)(i / Kpad), k = (int)(i % Kpad);
    out[i] = (k < K) ? (__bf16)w[(size_t)o * K + k] : (__bf16)0.0f;
  }
}

// [K][N] f32 (GCN weight) -> [N][K] bf16 (B-transposed layout)
__global__ void k_w_kn_to_nk_bf16(const float* __restrict__ w, __bf16* __restrict__ out,
                                  int K, int N) {
  size_t total = (size_t)K * N;
  size_t i = blockIdx.x * (size_t)blockDim.x + threadIdx.x;
  size_t st = (size_t)gridDim.x * blockDim.x;
  for (; i < total; i += st) {
    int nn = (int)(i / K), k = (int)(i % K);
    out[i] = (__bf16)w[(size_t)k * N + nn];
  }
}

// im2col with f32 -> bf16 conversion; output is [M][Kpad] row-major, K tail zeroed.
__global__ void k_im2col_bf16(const float* __restrict__ x, __bf16* __restrict__ col,
                              int Bn, int Cin, int H, int W, int KH, int KW,
                              int stride, int pad, int OH, int OW, int Kp) {
  size_t total = (size_t)Bn * OH * OW * Kp;
  size_t i = blockIdx.x * (size_t)blockDim.x + threadIdx.x;
  size_t st = (size_t)gridDim.x * blockDim.x;
  const int K = Cin * KH * KW;
  const int OHW = OH * OW;
  for (; i < total; i += st) {
    size_t m = i / Kp;
    int k = (int)(i - m * Kp);
    float v = 0.0f;
    if (k < K) {
      int ci = k / (KH * KW);
      int r  = k - ci * KH * KW;
      int kh = r / KW, kw = r - kh * KW;
      int img = (int)(m / OHW);
      int pix = (int)(m - (size_t)img * OHW);
      int oh = pix / OW, ow = pix - oh * OW;
      int ih = oh * stride - pad + kh;
      int iw = ow * stride - pad + kw;
      if (ih >= 0 && ih < H && iw >= 0 && iw < W)
        v = x[(((size_t)img * Cin + ci) * H + ih) * W + iw];
    }
    col[i] = (__bf16)v;
  }
}

// ---------------------------------------------------------------------------
// Register-blocked WMMA GEMM (conv path):
//   C[M,N] = A[M,Kp](bf16, row-major) * Bt[N,Kp](bf16, row-major)^T
// One wave computes a 32x64 output block: 2 M-tiles x 4 N-tiles of 16x16.
// Wave index comes through readfirstlane so tile guards are scalar branches
// (EXEC stays all-ones through every WMMA, as the ISA requires).
// ---------------------------------------------------------------------------
__global__ void k_wmma_gemm_bf16(const __bf16* __restrict__ A, const __bf16* __restrict__ Bt,
                                 float* __restrict__ C, int M, int N, int K,
                                 const float* __restrict__ gamma, const float* __restrict__ beta,
                                 const float* __restrict__ residual, int relu, int ohw) {
  int wave = __builtin_amdgcn_readfirstlane(
      (int)((blockIdx.x * blockDim.x + threadIdx.x) >> 5));
  int lane = threadIdx.x & 31;
  int mtiles = M >> 4;
  int nblk = N >> 6;               // groups of 4 n-tiles (N % 64 == 0 by construction)
  int mblk = (mtiles + 1) >> 1;    // groups of 2 m-tiles
  int wm = wave / nblk;
  int wn = wave - wm * nblk;
  if (wm >= mblk) return;          // scalar branch
  const int tm0  = wm * 2;
  const bool has2 = (tm0 + 1) < mtiles;

  const int ll = lane & 15;        // row (A) / col (B,C)
  const int hi = lane >> 4;        // K-half selector
  // A fragment: lane holds row ll, 16 contiguous K at hi*16
  const __bf16* arow0 = A + (size_t)(tm0 * 16 + ll) * K + hi * 16;
  const __bf16* arow1 = arow0 + (size_t)16 * K;
  // Bt fragment: lane holds column ll, 16 contiguous K at hi*16
  const __bf16* brow0 = Bt + (size_t)(wn * 64 + ll) * K + hi * 16;
  const size_t  bstep = (size_t)16 * K;   // next 16-column group

  v8f acc[8];
#pragma unroll
  for (int t = 0; t < 8; ++t)
#pragma unroll
    for (int r = 0; r < 8; ++r) acc[t][r] = 0.0f;

  for (int k0 = 0; k0 < K; k0 += 32) {
    __builtin_prefetch(arow0 + k0 + 512, 0, 0);   // global_prefetch_b8
    bf16x8 a0lo = *(const bf16x8*)(arow0 + k0);
    bf16x8 a0hi = *(const bf16x8*)(arow0 + k0 + 8);
    v16bf a0;
#pragma unroll
    for (int i = 0; i < 8; ++i) { a0[i] = a0lo[i]; a0[i + 8] = a0hi[i]; }

    v16bf bf[4];
#pragma unroll
    for (int j = 0; j < 4; ++j)
      bf[j] = *(const v16bf*)(brow0 + (size_t)j * bstep + k0);

#pragma unroll
    for (int j = 0; j < 4; ++j)
      acc[j] = __builtin_amdgcn_wmma_f32_16x16x32_bf16(
          false, a0, false, bf[j], (short)0, acc[j], false, false);

    if (has2) {
      bf16x8 a1lo = *(const bf16x8*)(arow1 + k0);
      bf16x8 a1hi = *(const bf16x8*)(arow1 + k0 + 8);
      v16bf a1;
#pragma unroll
      for (int i = 0; i < 8; ++i) { a1[i] = a1lo[i]; a1[i + 8] = a1hi[i]; }
#pragma unroll
      for (int j = 0; j < 4; ++j)
        acc[4 + j] = __builtin_amdgcn_wmma_f32_16x16x32_bf16(
            false, a1, false, bf[j], (short)0, acc[4 + j], false, false);
    }
  }

  // ---- epilogue ----
  float gv[4], bv[4];
#pragma unroll
  for (int j = 0; j < 4; ++j) {
    int n = wn * 64 + j * 16 + ll;
    gv[j] = gamma ? gamma[n] : 1.0f;
    bv[j] = gamma ? beta[n]  : 0.0f;
  }
#pragma unroll
  for (int t = 0; t < 8; ++t) {
    if (t >= 4 && !has2) continue;
    const int j  = t & 3;
    const int mt = tm0 + (t >> 2);
    const int n  = wn * 64 + j * 16 + ll;
    const int mbase = mt * 16 + hi * 8;
#pragma unroll
    for (int r = 0; r < 8; ++r) {
      int m = mbase + r;
      float v = acc[t][r] * gv[j] + bv[j];
      size_t idx;
      if (ohw > 0) {
        int img = m / ohw, pix = m - img * ohw;
        idx = ((size_t)img * N + n) * ohw + pix;
      } else {
        idx = (size_t)m * N + n;
      }
      if (residual) v += residual[idx];
      if (relu) v = fmaxf(v, 0.0f);
      C[idx] = v;
    }
  }
}

// ---------------------------------------------------------------------------
// GCN GEMM with TDM-staged B panel:
//   C[M,64] = A[M,64](bf16) * Bt[64,64](bf16)^T,  row-major f32 out.
// The whole 8 KB B^T panel is DMA'd into LDS once per workgroup by the
// Tensor Data Mover (tensor_load_to_lds, TENSORcnt-tracked). B fragments are
// then pulled from LDS with explicit ds_load_b128 inline asm — the TDM
// builtin's LDS write is invisible to alias analysis, so ordinary C++ loads
// of Bs get folded to poison; asm volatile loads cannot be.
// D# layout per CDNA5 ISA (2-byte elements, 1-D tile).
// ---------------------------------------------------------------------------
__global__ void __launch_bounds__(256)
k_wmma_gemm_gcn_tdm(const __bf16* __restrict__ A, const __bf16* __restrict__ Bt,
                    float* __restrict__ C, int M) {
  // 64 cols x 64 K of bf16 -> 8 KB staging panel in LDS.
  __shared__ v16bf Bs[256];
  // Generic VA low 32 bits == LDS byte offset (aperture translation, ISA 10.2);
  // also keeps the LDS allocation alive.
  const unsigned lds_base = (unsigned)(uintptr_t)Bs;

  if (threadIdx.x == 0) {          // TDM issues per-wave (EXEC ignored)
    unsigned long long ga = (unsigned long long)(uintptr_t)Bt;
    u32x4 g0 = { 1u,                                   // count=1, user descriptor
                 lds_base,                             // lds_addr (bytes)
                 (unsigned)ga,                         // global_addr[31:0]
                 (unsigned)((ga >> 32) & 0x01FFFFFFu)  // global_addr[56:32]
                   | (2u << 30) };                     // type=2 ("image")
    i32x8 g1 = { 0x00010000,   // [17:16] data_size=1 (2 bytes); wg_mask=0
                 0x10000000,   // tensor_dim0=4096 -> low16 at bits[31:16]
                 0x00010000,   // tensor_dim1=1   -> low16 at bits[31:16]
                 0x10000000,   // tile_dim0=4096  -> bits[31:16]
                 0,            // tile_dim1=0, tile_dim2=0 (1-D tile)
                 4096,         // tensor_dim0_stride = 4096 elements
                 0, 0 };       // tensor_dim1_stride = 0
    i32x4 gz  = { 0, 0, 0, 0 };                // groups 2/3 unused (<=2-D tensor)
    i32x8 gz8 = { 0, 0, 0, 0, 0, 0, 0, 0 };    // group 4 unused
    __builtin_amdgcn_tensor_load_to_lds(g0, g1, gz, gz, gz8, 0);
    __builtin_amdgcn_s_wait_tensorcnt(0);
  }
  __syncthreads();

  int wv = __builtin_amdgcn_readfirstlane(
      (int)(blockIdx.x * 8 + (threadIdx.x >> 5)));
  int lane = threadIdx.x & 31;
  int mtiles = M >> 4;
  int mblk = (mtiles + 1) >> 1;
  if (wv >= mblk) return;          // scalar branch (after the barrier)
  const int tm0  = wv * 2;
  const bool has2 = (tm0 + 1) < mtiles;

  const int ll = lane & 15;
  const int hi = lane >> 4;
  const __bf16* arow0 = A + (size_t)(tm0 * 16 + ll) * 64 + hi * 16;
  const __bf16* arow1 = arow0 + (size_t)16 * 64;

  v8f acc[8];
#pragma unroll
  for (int t = 0; t < 8; ++t)
#pragma unroll
    for (int r = 0; r < 8; ++r) acc[t][r] = 0.0f;

#pragma unroll
  for (int k0 = 0; k0 < 64; k0 += 32) {
    bf16x8 a0lo = *(const bf16x8*)(arow0 + k0);
    bf16x8 a0hi = *(const bf16x8*)(arow0 + k0 + 8);
    v16bf a0;
#pragma unroll
    for (int i = 0; i < 8; ++i) { a0[i] = a0lo[i]; a0[i + 8] = a0hi[i]; }

    v16bf bf[4];
#pragma unroll
    for (int j = 0; j < 4; ++j) {
      // B fragment: 32 contiguous bytes of LDS at column (j*16+ll), K = k0+hi*16
      int laddr = (int)lds_base + (((j * 16 + ll) * 64 + k0 + hi * 16) << 1);
      v4i qlo, qhi;
      asm volatile("ds_load_b128 %0, %2\n\t"
                   "ds_load_b128 %1, %2 offset:16\n\t"
                   "s_wait_dscnt 0x0"
                   : "=v"(qlo), "=v"(qhi)
                   : "v"(laddr));
      i32x8 wv8;
#pragma unroll
      for (int q = 0; q < 4; ++q) { wv8[q] = qlo[q]; wv8[q + 4] = qhi[q]; }
      bf[j] = __builtin_bit_cast(v16bf, wv8);
    }

#pragma unroll
    for (int j = 0; j < 4; ++j)
      acc[j] = __builtin_amdgcn_wmma_f32_16x16x32_bf16(
          false, a0, false, bf[j], (short)0, acc[j], false, false);

    if (has2) {
      bf16x8 a1lo = *(const bf16x8*)(arow1 + k0);
      bf16x8 a1hi = *(const bf16x8*)(arow1 + k0 + 8);
      v16bf a1;
#pragma unroll
      for (int i = 0; i < 8; ++i) { a1[i] = a1lo[i]; a1[i + 8] = a1hi[i]; }
#pragma unroll
      for (int j = 0; j < 4; ++j)
        acc[4 + j] = __builtin_amdgcn_wmma_f32_16x16x32_bf16(
            false, a1, false, bf[j], (short)0, acc[4 + j], false, false);
    }
  }

#pragma unroll
  for (int t = 0; t < 8; ++t) {
    if (t >= 4 && !has2) continue;
    const int j  = t & 3;
    const int mt = tm0 + (t >> 2);
    const int n  = j * 16 + ll;
    const int mbase = mt * 16 + hi * 8;
#pragma unroll
    for (int r = 0; r < 8; ++r)
      C[(size_t)(mbase + r) * 64 + n] = acc[t][r];
  }
}

// ---------------------------------------------------------------------------
// Pooling kernels
// ---------------------------------------------------------------------------
__global__ void k_maxpool3x3s2p1(const float* __restrict__ in, float* __restrict__ out,
                                 int Bn, int C, int H, int W, int OH, int OW) {
  size_t total = (size_t)Bn * C * OH * OW;
  size_t i = blockIdx.x * (size_t)blockDim.x + threadIdx.x;
  size_t st = (size_t)gridDim.x * blockDim.x;
  for (; i < total; i += st) {
    int ow = (int)(i % OW);
    size_t t = i / OW;
    int oh = (int)(t % OH);
    size_t bc = t / OH;
    const float* p = in + bc * (size_t)H * W;
    float m = -3.402823466e+38f;
    int h0 = oh * 2 - 1, w0 = ow * 2 - 1;
    for (int dh = 0; dh < 3; ++dh) {
      int ih = h0 + dh;
      if (ih < 0 || ih >= H) continue;
      for (int dw = 0; dw < 3; ++dw) {
        int iw = w0 + dw;
        if (iw < 0 || iw >= W) continue;
        m = fmaxf(m, p[(size_t)ih * W + iw]);
      }
    }
    out[i] = m;
  }
}

__global__ void k_avgpool(const float* __restrict__ in, float* __restrict__ out,
                          int BC, int HW) {
  int i = blockIdx.x * blockDim.x + threadIdx.x;
  if (i >= BC) return;
  const float* p = in + (size_t)i * HW;
  float s = 0.0f;
  for (int k = 0; k < HW; ++k) s += p[k];
  out[i] = s / (float)HW;
}

// ---------------------------------------------------------------------------
// GCN kernels
// ---------------------------------------------------------------------------
__global__ void k_edge_deg(const int* __restrict__ dst, long long E, float* __restrict__ deg) {
  long long i = blockIdx.x * (long long)blockDim.x + threadIdx.x;
  long long st = (long long)gridDim.x * blockDim.x;
  for (; i < E; i += st) atomicAdd(&deg[dst[i]], 1.0f);
}

__global__ void k_dinv(const float* __restrict__ deg, float* __restrict__ dinv, int n) {
  int i = blockIdx.x * blockDim.x + threadIdx.x;
  if (i < n) dinv[i] = rsqrtf(deg[i]);  // deg >= 1 (self-loop pre-seeded)
}

// aggf[i,f] = hw[i,f] * dinv[i]^2   (self-loop term; also initializes buffer)
__global__ void k_agg_init(const float* __restrict__ dinv, const float* __restrict__ hw,
                           float* __restrict__ out, size_t total, int F) {
  size_t i = blockIdx.x * (size_t)blockDim.x + threadIdx.x;
  size_t st = (size_t)gridDim.x * blockDim.x;
  for (; i < total; i += st) {
    float d = dinv[i / F];
    out[i] = hw[i] * d * d;
  }
}

__global__ void k_edge_scatter(const int* __restrict__ src, const int* __restrict__ dst,
                               const float* __restrict__ dinv, const float* __restrict__ hw,
                               float* __restrict__ out, long long E) {
  long long total = E * 16;  // 64 feats / 4 per thread
  long long i = blockIdx.x * (long long)blockDim.x + threadIdx.x;
  long long st = (long long)gridDim.x * blockDim.x;
  for (; i < total; i += st) {
    long long e = i >> 4;
    int q = (int)(i & 15);
    int s = src[e], d = dst[e];
    float c = dinv[s] * dinv[d];
    float4 v = ((const float4*)(hw + (size_t)s * 64))[q];
    float* o = out + (size_t)d * 64 + q * 4;
    atomicAdd(o + 0, v.x * c);
    atomicAdd(o + 1, v.y * c);
    atomicAdd(o + 2, v.z * c);
    atomicAdd(o + 3, v.w * c);
  }
}

__global__ void k_bias_relu(float* __restrict__ h, const float* __restrict__ bias,
                            size_t total, int F, int relu) {
  size_t i = blockIdx.x * (size_t)blockDim.x + threadIdx.x;
  size_t st = (size_t)gridDim.x * blockDim.x;
  for (; i < total; i += st) {
    float v = h[i] + bias[i % F];
    h[i] = relu ? fmaxf(v, 0.0f) : v;
  }
}

__global__ void k_pool_scatter(const float* __restrict__ h, const int* __restrict__ batch,
                               float* __restrict__ pooled, float* __restrict__ cnt, int n) {
  long long total = (long long)n * 16;
  long long i = blockIdx.x * (long long)blockDim.x + threadIdx.x;
  long long st = (long long)gridDim.x * blockDim.x;
  for (; i < total; i += st) {
    int node = (int)(i >> 4);
    int q = (int)(i & 15);
    int b = batch[node];
    float4 v = ((const float4*)(h + (size_t)node * 64))[q];
    float* o = pooled + (size_t)b * 64 + q * 4;
    atomicAdd(o + 0, v.x);
    atomicAdd(o + 1, v.y);
    atomicAdd(o + 2, v.z);
    atomicAdd(o + 3, v.w);
    if (q == 0) atomicAdd(&cnt[b], 1.0f);
  }
}

__global__ void k_pool_div(float* __restrict__ pooled, const float* __restrict__ cnt, int total) {
  int i = blockIdx.x * blockDim.x + threadIdx.x;
  if (i < total) pooled[i] /= fmaxf(cnt[i / 64], 1.0f);
}

__global__ void k_final_linear(const float* __restrict__ pooled, const float* __restrict__ feats,
                               const float* __restrict__ Wl, const float* __restrict__ bl,
                               float* __restrict__ out, int Bn, int ncls) {
  int t = blockIdx.x * blockDim.x + threadIdx.x;
  if (t >= Bn * ncls) return;
  int b = t / ncls, c = t - b * ncls;
  float acc = bl[c];
  for (int k = 0; k < 64; ++k)  acc += pooled[(size_t)b * 64 + k]  * Wl[(size_t)k * ncls + c];
  for (int k = 0; k < 512; ++k) acc += feats[(size_t)b * 512 + k] * Wl[(size_t)(64 + k) * ncls + c];
  out[t] = acc;
}

// ---------------------------------------------------------------------------
// Host-side helpers (stateless, graph-capture safe: only kernel launches)
// ---------------------------------------------------------------------------
static inline unsigned gsz(size_t n, int tpb) {
  size_t b = (n + (size_t)tpb - 1) / tpb;
  if (b == 0) b = 1;
  if (b > 1048576) b = 1048576;
  return (unsigned)b;
}

static void launch_gemm(hipStream_t stream, const __bf16* A, const __bf16* Bt, float* C,
                        int M, int N, int K, const float* g, const float* b,
                        const float* residual, int relu, int ohw) {
  size_t mblk = ((size_t)(M / 16) + 1) / 2;
  size_t nblk = (size_t)N / 64;
  size_t threads = mblk * nblk * 32;
  k_wmma_gemm_bf16<<<(unsigned)((threads + 255) / 256), 256, 0, stream>>>(
      A, Bt, C, M, N, K, g, b, residual, relu, ohw);
}

static void conv_bn(hipStream_t stream, const float* in, float* out, const __bf16* wbf,
                    int Bn, int Cin, int H, int W, int Cout, int KH, int KW,
                    int stride, int pad, const float* g, const float* b,
                    const float* residual, int relu, __bf16* colbuf) {
  int OH = (H + 2 * pad - KH) / stride + 1;
  int OW = (W + 2 * pad - KW) / stride + 1;
  int K  = Cin * KH * KW;
  int Kp = (K + 31) & ~31;
  size_t M = (size_t)Bn * OH * OW;
  k_im2col_bf16<<<gsz(M * (size_t)Kp, 256), 256, 0, stream>>>(
      in, colbuf, Bn, Cin, H, W, KH, KW, stride, pad, OH, OW, Kp);
  launch_gemm(stream, colbuf, wbf, out, (int)M, Cout, Kp, g, b, residual, relu, OH * OW);
}

extern "C" void kernel_launch(void* const* d_in, const int* in_sizes, int n_in,
                              void* d_out, int out_size, void* d_ws, size_t ws_size,
                              hipStream_t stream) {
  (void)n_in; (void)out_size; (void)ws_size;
  const float* x     = (const float*)d_in[0];
  const int*   eidx  = (const int*)d_in[1];
  const float* ximg  = (const float*)d_in[2];
  const int*   batch = (const int*)d_in[3];
  const float* W1 = (const float*)d_in[4];  const float* b1 = (const float*)d_in[5];
  const float* W2 = (const float*)d_in[6];  const float* b2 = (const float*)d_in[7];
  const float* W3 = (const float*)d_in[8];  const float* b3 = (const float*)d_in[9];
  const float* linW = (const float*)d_in[10]; const float* linb = (const float*)d_in[11];

  const int       Nn   = in_sizes[0] / 64;
  const long long E    = in_sizes[1] / 2;
  const int       Bimg = in_sizes[2] / (3 * 224 * 224);
  const int       NCLS = 10;

  // ---- resnet_params leaf index map (JAX pytree order: blocks list, sorted dict keys,
  //      then stem_b/stem_g/stem_w) ----
  const int cinA[8]  = {64, 64, 64, 128, 128, 256, 256, 512};
  const int coutA[8] = {64, 64, 128, 128, 256, 256, 512, 512};
  const int strA[8]  = {1, 1, 2, 1, 2, 1, 2, 1};
  const int hasD[8]  = {0, 0, 1, 0, 1, 0, 1, 0};
  int b1i[8], b2i[8], g1i[8], g2i[8], w1i[8], w2i[8], bdi[8], gdi[8], wdi[8];
  int stem_b_i, stem_g_i, stem_w_i;
  {
    int c = 12;
    for (int k = 0; k < 8; ++k) {
      if (!hasD[k]) {
        b1i[k]=c; b2i[k]=c+1; g1i[k]=c+2; g2i[k]=c+3; w1i[k]=c+4; w2i[k]=c+5; c += 6;
        bdi[k]=gdi[k]=wdi[k]=-1;
      } else {
        b1i[k]=c; b2i[k]=c+1; bdi[k]=c+2; g1i[k]=c+3; g2i[k]=c+4; gdi[k]=c+5;
        w1i[k]=c+6; w2i[k]=c+7; wdi[k]=c+8; c += 9;
      }
    }
    stem_b_i = c; stem_g_i = c + 1; stem_w_i = c + 2;
  }

  // ---- workspace carve ----
  uintptr_t cur = (uintptr_t)d_ws;
  auto alloc = [&](size_t bytes) -> void* {
    void* p = (void*)cur;
    cur = (cur + bytes + 255) & ~(uintptr_t)255;
    return p;
  };

  __bf16* xbf    = (__bf16*)alloc((size_t)Nn * 64 * sizeof(__bf16));
  float*  hwf    = (float*)alloc((size_t)Nn * 64 * sizeof(float));
  float*  aggf   = (float*)alloc((size_t)Nn * 64 * sizeof(float));
  float*  deg    = (float*)alloc((size_t)Nn * sizeof(float));
  float*  dinvv  = (float*)alloc((size_t)Nn * sizeof(float));
  float*  pooled = (float*)alloc((size_t)Bimg * 64 * sizeof(float));
  float*  cnt    = (float*)alloc((size_t)Bimg * sizeof(float));
  __bf16* wt[3];
  for (int i = 0; i < 3; ++i) wt[i] = (__bf16*)alloc(64 * 64 * sizeof(__bf16));

  __bf16* colbuf  = (__bf16*)alloc((size_t)Bimg * 112 * 112 * 160 * sizeof(__bf16));
  float*  stemout = (float*)alloc((size_t)Bimg * 64 * 112 * 112 * sizeof(float));
  float*  acts[4];
  for (int i = 0; i < 4; ++i) acts[i] = (float*)alloc((size_t)Bimg * 64 * 56 * 56 * sizeof(float));
  float*  feats = (float*)alloc((size_t)Bimg * 512 * sizeof(float));

  // ---- ResNet weights: f32 OIHW -> bf16 [O][Kpad] ----
  __bf16* stemw = (__bf16*)alloc((size_t)64 * 160 * sizeof(__bf16));
  k_w_pad_bf16<<<gsz((size_t)64 * 160, 256), 256, 0, stream>>>(
      (const float*)d_in[stem_w_i], stemw, 64, 147, 160);
  __bf16 *w1b[8], *w2b[8], *wdb[8];
  for (int k = 0; k < 8; ++k) {
    int K1 = cinA[k] * 9,  Kp1 = (K1 + 31) & ~31;
    int K2 = coutA[k] * 9, Kp2 = (K2 + 31) & ~31;
    w1b[k] = (__bf16*)alloc((size_t)coutA[k] * Kp1 * sizeof(__bf16));
    k_w_pad_bf16<<<gsz((size_t)coutA[k] * Kp1, 256), 256, 0, stream>>>(
        (const float*)d_in[w1i[k]], w1b[k], coutA[k], K1, Kp1);
    w2b[k] = (__bf16*)alloc((size_t)coutA[k] * Kp2 * sizeof(__bf16));
    k_w_pad_bf16<<<gsz((size_t)coutA[k] * Kp2, 256), 256, 0, stream>>>(
        (const float*)d_in[w2i[k]], w2b[k], coutA[k], K2, Kp2);
    if (hasD[k]) {
      int Kd = cinA[k], Kpd = (Kd + 31) & ~31;
      wdb[k] = (__bf16*)alloc((size_t)coutA[k] * Kpd * sizeof(__bf16));
      k_w_pad_bf16<<<gsz((size_t)coutA[k] * Kpd, 256), 256, 0, stream>>>(
          (const float*)d_in[wdi[k]], wdb[k], coutA[k], Kd, Kpd);
    } else {
      wdb[k] = nullptr;
    }
  }

  // =========================== GCN branch ===========================
  const int* src = eidx;
  const int* dst = eidx + E;

  k_f32_to_bf16<<<gsz((size_t)Nn * 64, 256), 256, 0, stream>>>(x, xbf, (size_t)Nn * 64);
  k_w_kn_to_nk_bf16<<<gsz(64 * 64, 256), 256, 0, stream>>>(W1, wt[0], 64, 64);
  k_w_kn_to_nk_bf16<<<gsz(64 * 64, 256), 256, 0, stream>>>(W2, wt[1], 64, 64);
  k_w_kn_to_nk_bf16<<<gsz(64 * 64, 256), 256, 0, stream>>>(W3, wt[2], 64, 64);

  k_fill_f32<<<gsz(Nn, 256), 256, 0, stream>>>(deg, (size_t)Nn, 1.0f);  // self-loop
  k_edge_deg<<<gsz((size_t)E, 256), 256, 0, stream>>>(dst, E, deg);
  k_dinv<<<gsz(Nn, 256), 256, 0, stream>>>(deg, dinvv, Nn);

  const float* biases[3] = {b1, b2, b3};
  const int mblk_gcn = ((Nn / 16) + 1) / 2;
  const unsigned gcn_blocks = (unsigned)((mblk_gcn + 7) / 8);
  for (int l = 0; l < 3; ++l) {
    // dense: hwf = H_in(bf16) @ W_l  (WMMA + TDM-staged B panel in LDS)
    k_wmma_gemm_gcn_tdm<<<gcn_blocks, 256, 0, stream>>>(xbf, wt[l], hwf, Nn);
    // aggregate: self-loop init + edge scatter-add
    k_agg_init<<<gsz((size_t)Nn * 64, 256), 256, 0, stream>>>(
        dinvv, hwf, aggf, (size_t)Nn * 64, 64);
    k_edge_scatter<<<gsz((size_t)E * 16, 256), 256, 0, stream>>>(
        src, dst, dinvv, hwf, aggf, E);
    k_bias_relu<<<gsz((size_t)Nn * 64, 256), 256, 0, stream>>>(
        aggf, biases[l], (size_t)Nn * 64, 64, l < 2 ? 1 : 0);
    if (l < 2)
      k_f32_to_bf16<<<gsz((size_t)Nn * 64, 256), 256, 0, stream>>>(aggf, xbf, (size_t)Nn * 64);
  }

  k_fill_f32<<<gsz((size_t)Bimg * 64, 256), 256, 0, stream>>>(pooled, (size_t)Bimg * 64, 0.0f);
  k_fill_f32<<<gsz(Bimg, 256), 256, 0, stream>>>(cnt, (size_t)Bimg, 0.0f);
  k_pool_scatter<<<gsz((size_t)Nn * 16, 256), 256, 0, stream>>>(aggf, batch, pooled, cnt, Nn);
  k_pool_div<<<gsz(Bimg * 64, 256), 256, 0, stream>>>(pooled, cnt, Bimg * 64);

  // =========================== ResNet-18 branch ===========================
  // stem: 7x7 s2 p3 conv + BN + ReLU -> [B,64,112,112]
  conv_bn(stream, ximg, stemout, stemw, Bimg, 3, 224, 224, 64, 7, 7, 2, 3,
          (const float*)d_in[stem_g_i], (const float*)d_in[stem_b_i], nullptr, 1, colbuf);
  // maxpool 3x3 s2 p1 -> [B,64,56,56]
  float* X  = acts[0];
  float* T1 = acts[1];
  float* T2 = acts[2];
  float* Dn = acts[3];
  k_maxpool3x3s2p1<<<gsz((size_t)Bimg * 64 * 56 * 56, 256), 256, 0, stream>>>(
      stemout, X, Bimg, 64, 112, 112, 56, 56);

  int Hc = 56;
  for (int k = 0; k < 8; ++k) {
    int cin = cinA[k], cout = coutA[k], s = strA[k];
    int OHc = (Hc + 2 - 3) / s + 1;
    conv_bn(stream, X, T1, w1b[k], Bimg, cin, Hc, Hc, cout, 3, 3, s, 1,
            (const float*)d_in[g1i[k]], (const float*)d_in[b1i[k]], nullptr, 1, colbuf);
    const float* idn = X;
    if (hasD[k]) {
      conv_bn(stream, X, Dn, wdb[k], Bimg, cin, Hc, Hc, cout, 1, 1, s, 0,
              (const float*)d_in[gdi[k]], (const float*)d_in[bdi[k]], nullptr, 0, colbuf);
      idn = Dn;
    }
    conv_bn(stream, T1, T2, w2b[k], Bimg, cout, OHc, OHc, cout, 3, 3, 1, 1,
            (const float*)d_in[g2i[k]], (const float*)d_in[b2i[k]], idn, 1, colbuf);
    float* t = X; X = T2; T2 = t;  // rotate: block output becomes next input
    Hc = OHc;
  }
  // global avg pool: [B,512,7,7] -> [B,512]
  k_avgpool<<<gsz((size_t)Bimg * 512, 256), 256, 0, stream>>>(X, feats, Bimg * 512, 49);

  // =========================== final linear ===========================
  k_final_linear<<<gsz((size_t)Bimg * NCLS, 256), 256, 0, stream>>>(
      pooled, feats, linW, linb, (float*)d_out, Bimg, NCLS);
}
